// RNN_51805895524907
// MI455X (gfx1250) — matrix operand order repaired and checked by reading, not compiled, not run on hardware
//
#include <hip/hip_runtime.h>

// ---------------------------------------------------------------------------
// CDNA5 / gfx1250 2-layer Elman RNN using v_wmma_f32_16x16x32_bf16.
//   - Parallel pre-GEMM:  xw0[t,b,:] = x[b,t,:] @ W_ih0^T + b_ih0 + b_hh0  (bf16)
//   - Recurrence pipelined as:  step0(0) ; { step1(t) || step0(t+1) } x511 ;
//     step1(511)  -- halves sequential launch count (launch latency bound).
//   - Final FC:           out = h1 @ fcW^T + fcb                           (f32)
// Weights pre-packed into WMMA B-fragment layout (lane%16 = N col, 16 contiguous
// bf16 per lane) -> 8 MB bf16 resident in L2 for all 512 steps.  Hidden states
// stored in WMMA A-fragment packed layout; each 8-wave workgroup stages its
// 32 KB A panel into LDS once and all waves feed WMMA from ds_load (8x cut in
// L2 A-traffic on the sequential critical path).
// ---------------------------------------------------------------------------

typedef __attribute__((ext_vector_type(16))) __bf16 v16bf;
typedef __attribute__((ext_vector_type(8)))  float  v8f;

#define WMMA_BF16(a, b, c) \
  __builtin_amdgcn_wmma_f32_16x16x32_bf16(false, (a), false, (b), (short)0, (c), false, false)

constexpr int Bc = 64, Sc = 512, Dc = 1024, Hc = 1024, Oc = 512;
constexpr int KT_H = Hc / 32;       // 32 K-tiles over H
constexpr int KT_D = Dc / 32;       // 32 K-tiles over D
constexpr int APANEL = 16 * Hc;     // one mt's packed A panel: 16384 bf16 = 32 KB

// ---- pack f32 row-major W[N][K] -> bf16 B-fragment tiles -------------------
// out index = ((nt*Kt + kt)*32 + lane)*16 + i ;  lane&15 -> N col, (lane>>4, i) -> K
__global__ void rnn_pack_w(const float* __restrict__ W, __bf16* __restrict__ out,
                           int N, int K) {
  int idx = blockIdx.x * 256 + threadIdx.x;
  if (idx >= N * K) return;
  int i    = idx & 15;
  int lane = (idx >> 4) & 31;
  int tile = idx >> 9;
  int Kt   = K >> 5;
  int kt   = tile % Kt;
  int nt   = tile / Kt;
  int n = nt * 16 + (lane & 15);
  int k = kt * 32 + ((lane >> 4) << 3) + (i & 7) + ((i & 8) << 1);
  out[idx] = (__bf16)W[(size_t)n * K + k];
}

__global__ void rnn_zero(__bf16* __restrict__ p, int n) {
  int i = blockIdx.x * 256 + threadIdx.x;
  if (i < n) p[i] = (__bf16)0.0f;
}

// ---- parallel pre-GEMM: xw0 = x @ W_ih0^T + b_ih0 + b_hh0 ------------------
// Rows R = t*B + b (so xw0 is [S][B][H]); A row R comes from x[b][t][:].
// One wave computes one 16-row M-tile x two 16-col N-tiles.  HBM-bound on x.
__global__ void rnn_xw0_gemm(const float* __restrict__ x,
                             const __bf16* __restrict__ wp,
                             const float* __restrict__ bi,
                             const float* __restrict__ bh,
                             __bf16* __restrict__ xw0) {
  const int lane = threadIdx.x & 31;
  const int wave = threadIdx.x >> 5;
  const int job  = blockIdx.x * 8 + wave;          // 65536 jobs
  const int ntp  = job & 31;                       // n-tile pair
  const int mt   = job >> 5;                       // 0..2047
  const int m    = mt * 16 + (lane & 15);
  const int bb   = m & (Bc - 1);
  const int tt   = m >> 6;
  // A-fragment K offset: lanes>=16 handle K%32 in {8..15, 24..31}
  const float* arow = x + ((size_t)bb * Sc + tt) * Dc + ((lane >> 4) << 3);
  const __bf16* bp0 = wp + (((size_t)(2 * ntp)     * KT_D) << 9);
  const __bf16* bp1 = wp + (((size_t)(2 * ntp + 1) * KT_D) << 9);
  v8f c0 = {}; v8f c1 = {};
  for (int kt = 0; kt < KT_D; ++kt) {
    const float* p = arow + kt * 32;
    v16bf a;
#pragma unroll
    for (int j = 0; j < 4; ++j) {             // elems 0..7 -> K+0..7, 8..15 -> K+16..23
      float2 f0 = *(const float2*)(p + 2 * j);
      float2 f1 = *(const float2*)(p + 16 + 2 * j);
      a[2 * j]     = (__bf16)f0.x; a[2 * j + 1]     = (__bf16)f0.y;
      a[8 + 2 * j] = (__bf16)f1.x; a[8 + 2 * j + 1] = (__bf16)f1.y;
    }
    v16bf b0 = *(const v16bf*)(bp0 + (((size_t)kt * 32 + lane) << 4));
    v16bf b1 = *(const v16bf*)(bp1 + (((size_t)kt * 32 + lane) << 4));
    c0 = WMMA_BF16(a, b0, c0);
    c1 = WMMA_BF16(a, b1, c1);
  }
  const int n0 = (2 * ntp)     * 16 + (lane & 15);
  const int n1 = (2 * ntp + 1) * 16 + (lane & 15);
  const float bias0 = bi[n0] + bh[n0];
  const float bias1 = bi[n1] + bh[n1];
#pragma unroll
  for (int r = 0; r < 8; ++r) {              // D layout: lanes>=16 -> M += 8
    const int Mr = mt * 16 + r + ((lane >> 4) << 3);
    xw0[(size_t)Mr * Hc + n0] = (__bf16)(c0[r] + bias0);
    xw0[(size_t)Mr * Hc + n1] = (__bf16)(c1[r] + bias1);
  }
}

// ---- cooperative A-panel stage: 32 KB global -> LDS with 256 threads -------
__device__ __forceinline__ void load_a_to_lds(__bf16* dst, const __bf16* src,
                                              int tid) {
  // 16384 bf16 = 1024 chunks of v16bf (32 B); 4 chunks per thread.
#pragma unroll
  for (int j = 0; j < 4; ++j) {
    ((v16bf*)dst)[j * 256 + tid] = ((const v16bf*)src)[j * 256 + tid];
  }
}

// D-element -> A-fragment packed-store index math (branchless):
//   for output column n (next GEMM's K): kk = n%32, kt2 = n/32,
//   ii = (kk&7) | ((kk&16)>>1), lane' = (M&15) + ((kk&8)<<1)

// ---- layer-0 step body: h0 = tanh(xw_t + h_prev @ W_hh0^T) -----------------
// a_lds: this mt's 32 KB A panel already staged in LDS.
__device__ __forceinline__ void step0_body(const __bf16* __restrict__ a_lds,
                                           const __bf16* __restrict__ whh,
                                           const __bf16* __restrict__ xw_t,
                                           __bf16* __restrict__ h_cur,
                                           int mt, int nt, int lane) {
  const __bf16* bp = whh + (((size_t)nt * KT_H) << 9);
  v8f c = {};
#pragma unroll 4
  for (int kt = 0; kt < KT_H; ++kt) {
    const size_t o = ((size_t)kt * 32 + lane) << 4;
    v16bf a = *(const v16bf*)(a_lds + o);
    v16bf b = *(const v16bf*)(bp + o);
    c = WMMA_BF16(a, b, c);
  }
  const int n     = nt * 16 + (lane & 15);
  const int kk    = n & 31;
  const int kt2   = n >> 5;
  const int ii    = (kk & 7) | ((kk & 16) >> 1);
  const int lpadd = (kk & 8) << 1;
#pragma unroll
  for (int r = 0; r < 8; ++r) {
    const int Mr = mt * 16 + r + ((lane >> 4) << 3);
    float v = tanhf(c[r] + (float)xw_t[(size_t)Mr * Hc + n]);
    const int lp = (Mr & 15) + lpadd;
    h_cur[((((size_t)(Mr >> 4) * KT_H + kt2) * 32 + lp) << 4) + ii] = (__bf16)v;
  }
}

// ---- layer-1 step body: h1 = tanh(h0 @ W_ih1^T + h1_prev @ W_hh1^T + b) ----
__device__ __forceinline__ void step1_body(const __bf16* __restrict__ a0_lds,
                                           const __bf16* __restrict__ a1_lds,
                                           const __bf16* __restrict__ wih,
                                           const __bf16* __restrict__ whh,
                                           const float* __restrict__ bi,
                                           const float* __restrict__ bh,
                                           __bf16* __restrict__ h1_cur,
                                           int mt, int nt, int lane) {
  const __bf16* b0p = wih + (((size_t)nt * KT_H) << 9);
  const __bf16* b1p = whh + (((size_t)nt * KT_H) << 9);
  v8f c = {};
#pragma unroll 4
  for (int kt = 0; kt < KT_H; ++kt) {
    const size_t o = ((size_t)kt * 32 + lane) << 4;
    v16bf a0 = *(const v16bf*)(a0_lds + o);
    v16bf b0 = *(const v16bf*)(b0p + o);
    c = WMMA_BF16(a0, b0, c);
    v16bf a1 = *(const v16bf*)(a1_lds + o);
    v16bf b1 = *(const v16bf*)(b1p + o);
    c = WMMA_BF16(a1, b1, c);
  }
  const int n     = nt * 16 + (lane & 15);
  const float bias = bi[n] + bh[n];
  const int kk    = n & 31;
  const int kt2   = n >> 5;
  const int ii    = (kk & 7) | ((kk & 16) >> 1);
  const int lpadd = (kk & 8) << 1;
#pragma unroll
  for (int r = 0; r < 8; ++r) {
    const int Mr = mt * 16 + r + ((lane >> 4) << 3);
    float v = tanhf(c[r] + bias);
    const int lp = (Mr & 15) + lpadd;
    h1_cur[((((size_t)(Mr >> 4) * KT_H + kt2) * 32 + lp) << 4) + ii] = (__bf16)v;
  }
}

// ---- standalone step kernels (pipeline prologue / epilogue) ----------------
// 32 blocks x 8 waves; block owns one mt (A panel in LDS) and 8 n-tiles.
__global__ void rnn_step0(const __bf16* __restrict__ h_prev,
                          const __bf16* __restrict__ whh,
                          const __bf16* __restrict__ xw_t,
                          __bf16* __restrict__ h_cur) {
  __shared__ alignas(32) __bf16 lds_a[APANEL];
  const int lane = threadIdx.x & 31;
  const int wave = threadIdx.x >> 5;
  const int mt = blockIdx.x & 3;
  const int nt = (blockIdx.x >> 2) * 8 + wave;
  load_a_to_lds(lds_a, h_prev + (((size_t)mt * KT_H) << 9), threadIdx.x);
  __syncthreads();
  step0_body(lds_a, whh, xw_t, h_cur, mt, nt, lane);
}

__global__ void rnn_step1(const __bf16* __restrict__ h0_cur,
                          const __bf16* __restrict__ h1_prev,
                          const __bf16* __restrict__ wih,
                          const __bf16* __restrict__ whh,
                          const float* __restrict__ bi,
                          const float* __restrict__ bh,
                          __bf16* __restrict__ h1_cur) {
  __shared__ alignas(32) __bf16 lds_a[2][APANEL];
  const int lane = threadIdx.x & 31;
  const int wave = threadIdx.x >> 5;
  const int mt = blockIdx.x & 3;
  const int nt = (blockIdx.x >> 2) * 8 + wave;
  load_a_to_lds(lds_a[0], h0_cur  + (((size_t)mt * KT_H) << 9), threadIdx.x);
  load_a_to_lds(lds_a[1], h1_prev + (((size_t)mt * KT_H) << 9), threadIdx.x);
  __syncthreads();
  step1_body(lds_a[0], lds_a[1], wih, whh, bi, bh, h1_cur, mt, nt, lane);
}

// ---- fused pipeline stage: { step1(t) || step0(t+1) } in one launch --------
// step1(t) needs h0(t), h1(t-1);  step0(t+1) needs h0(t), xw0[t+1].
// Both only consume data produced by the PREVIOUS launch -> safe to co-issue.
// The layer split is block-uniform (blockIdx), so barriers and WMMA EXEC are
// safe; blocks 0..31 run layer1(t), blocks 32..63 run layer0(t+1).
__global__ void rnn_fused(const __bf16* __restrict__ h0_cur,   // packed A, h0(t)
                          const __bf16* __restrict__ h1_prev,  // packed A, h1(t-1)
                          const __bf16* __restrict__ whh0,
                          const __bf16* __restrict__ wih1,
                          const __bf16* __restrict__ whh1,
                          const __bf16* __restrict__ xw_next,  // xw0[t+1]
                          const float* __restrict__ bi1,
                          const float* __restrict__ bh1,
                          __bf16* __restrict__ h1_out,         // h1(t)
                          __bf16* __restrict__ h0_next) {      // h0(t+1)
  __shared__ alignas(32) __bf16 lds_a[2][APANEL];
  const int lane = threadIdx.x & 31;
  const int wave = threadIdx.x >> 5;
  const int bb = blockIdx.x & 31;
  const int mt = bb & 3;
  const int nt = (bb >> 2) * 8 + wave;
  if (blockIdx.x < 32) {
    load_a_to_lds(lds_a[0], h0_cur  + (((size_t)mt * KT_H) << 9), threadIdx.x);
    load_a_to_lds(lds_a[1], h1_prev + (((size_t)mt * KT_H) << 9), threadIdx.x);
    __syncthreads();
    step1_body(lds_a[0], lds_a[1], wih1, whh1, bi1, bh1, h1_out, mt, nt, lane);
  } else {
    load_a_to_lds(lds_a[0], h0_cur + (((size_t)mt * KT_H) << 9), threadIdx.x);
    __syncthreads();
    step0_body(lds_a[0], whh0, xw_next, h0_next, mt, nt, lane);
  }
}

// ---- final FC: out = h1 @ fcW^T + fcb  (f32 row-major output) --------------
__global__ void rnn_fc(const __bf16* __restrict__ h1,   // packed A
                       const __bf16* __restrict__ wp,   // packed B (Oc x Hc)
                       const float* __restrict__ bias,
                       float* __restrict__ out) {
  const int lane = threadIdx.x & 31;
  const int wave = threadIdx.x >> 5;
  const int tile = blockIdx.x * 8 + wave;    // 128 tiles (4 mt x 32 nt)
  const int nt = tile & 31;
  const int mt = tile >> 5;
  const __bf16* ap = h1 + (((size_t)mt * KT_H) << 9);
  const __bf16* bp = wp + (((size_t)nt * KT_H) << 9);
  v8f c = {};
#pragma unroll 4
  for (int kt = 0; kt < KT_H; ++kt) {
    const size_t o = ((size_t)kt * 32 + lane) << 4;
    v16bf a = *(const v16bf*)(ap + o);
    v16bf b = *(const v16bf*)(bp + o);
    c = WMMA_BF16(a, b, c);
  }
  const int n = nt * 16 + (lane & 15);
  const float bv = bias[n];
#pragma unroll
  for (int r = 0; r < 8; ++r) {
    const int Mr = mt * 16 + r + ((lane >> 4) << 3);
    out[(size_t)Mr * Oc + n] = c[r] + bv;
  }
}

// ---------------------------------------------------------------------------
extern "C" void kernel_launch(void* const* d_in, const int* in_sizes, int n_in,
                              void* d_out, int out_size, void* d_ws, size_t ws_size,
                              hipStream_t stream) {
  (void)in_sizes; (void)n_in; (void)out_size; (void)ws_size;
  const float* x     = (const float*)d_in[0];
  const float* W_ih0 = (const float*)d_in[1];
  const float* W_hh0 = (const float*)d_in[2];
  const float* b_ih0 = (const float*)d_in[3];
  const float* b_hh0 = (const float*)d_in[4];
  const float* W_ih1 = (const float*)d_in[5];
  const float* W_hh1 = (const float*)d_in[6];
  const float* b_ih1 = (const float*)d_in[7];
  const float* b_hh1 = (const float*)d_in[8];
  const float* fcW   = (const float*)d_in[9];
  const float* fcb   = (const float*)d_in[10];

  char* ws = (char*)d_ws;
  const size_t MB = 1024 * 1024;
  __bf16* wih0p = (__bf16*)(ws + 0 * MB);   // 2 MB each (1024x1024 bf16)
  __bf16* whh0p = (__bf16*)(ws + 2 * MB);
  __bf16* wih1p = (__bf16*)(ws + 4 * MB);
  __bf16* whh1p = (__bf16*)(ws + 6 * MB);
  __bf16* fcwp  = (__bf16*)(ws + 8 * MB);   // 1 MB (512x1024 bf16)
  __bf16* xw0   = (__bf16*)(ws + 16 * MB);  // 64 MB: [S][B][H] bf16
  __bf16* hbase = (__bf16*)(ws + 80 * MB);  // 4 packed h buffers, 128 KB each
  __bf16* h0p[2] = { hbase,                  hbase + (size_t)Bc * Hc };
  __bf16* h1p[2] = { hbase + 2ull * Bc * Hc, hbase + 3ull * Bc * Hc };

  // Pack weights to bf16 WMMA-B layout (weights then live in L2 for all steps).
  rnn_pack_w<<<(Hc * Dc) / 256, 256, 0, stream>>>(W_ih0, wih0p, Hc, Dc);
  rnn_pack_w<<<(Hc * Hc) / 256, 256, 0, stream>>>(W_hh0, whh0p, Hc, Hc);
  rnn_pack_w<<<(Hc * Hc) / 256, 256, 0, stream>>>(W_ih1, wih1p, Hc, Hc);
  rnn_pack_w<<<(Hc * Hc) / 256, 256, 0, stream>>>(W_hh1, whh1p, Hc, Hc);
  rnn_pack_w<<<(Oc * Hc) / 256, 256, 0, stream>>>(fcW,   fcwp,  Oc, Hc);

  // Zero initial hidden states (both parities; deterministic per launch).
  rnn_zero<<<(4 * Bc * Hc) / 256, 256, 0, stream>>>(hbase, 4 * Bc * Hc);

  // Parallel hoisted input GEMM (M=32768, N=1024, K=1024).
  rnn_xw0_gemm<<<8192, 256, 0, stream>>>(x, wih0p, b_ih0, b_hh0, xw0);

  // Pipelined recurrence: 514 sequential launches instead of 1025.
  //   prologue: h0(0) = tanh(xw0[0] + 0 @ Whh0)     (reads zeroed h0p[1])
  rnn_step0<<<32, 256, 0, stream>>>(h0p[1], whh0p, xw0, h0p[0]);
  //   steady state: fused(t) = { step1(t) , step0(t+1) },  t = 0..510
  for (int t = 0; t < Sc - 1; ++t) {
    const int p = t & 1;
    rnn_fused<<<64, 256, 0, stream>>>(h0p[p], h1p[p ^ 1], whh0p, wih1p, whh1p,
                                      xw0 + (size_t)(t + 1) * Bc * Hc,
                                      b_ih1, b_hh1, h1p[p], h0p[p ^ 1]);
  }
  //   epilogue: step1(511)  (p = 1)
  rnn_step1<<<32, 256, 0, stream>>>(h0p[1], h1p[0], wih1p, whh1p,
                                    b_ih1, b_hh1, h1p[1]);

  rnn_fc<<<16, 256, 0, stream>>>(h1p[1], fcwp, fcb, (float*)d_out);
}
